// MLPDecoder_15307263443115
// MI455X (gfx1250) — matrix operand order repaired and checked by reading, not compiled, not run on hardware
//
#include <hip/hip_runtime.h>
#include <hip/hip_bf16.h>

typedef __attribute__((ext_vector_type(16))) __bf16 v16bf;
typedef __attribute__((ext_vector_type(8)))  float  v8f;

#define C_N 512
#define D_N 256
#define H_N 512
#define M_N 256
#define T_N 2
#define E_N (C_N * (C_N - 1))   // 261632

__device__ __forceinline__ unsigned short f2bf(float x) {
    unsigned u = __float_as_uint(x);
    u += 0x7FFFu + ((u >> 16) & 1u);   // round-to-nearest-even
    return (unsigned short)(u >> 16);
}
__device__ __forceinline__ float bf2f(unsigned short b) {
    return __uint_as_float(((unsigned)b) << 16);
}

// ---------------------------------------------------------------------------
// Kernel 1: S1[t][c][h] = bf16( b1[t][h] + inputs[c] . W1[t][:D][:,h] )
//           R1[t][c][h] = bf16(            inputs[c] . W1[t][D:][:,h] )
// grid = T*C*2 blocks of 256 threads (h split in two halves)
// ---------------------------------------------------------------------------
__global__ void precompute_node_terms(const float* __restrict__ inputs,
                                      const float* __restrict__ W1,
                                      const float* __restrict__ b1,
                                      unsigned short* __restrict__ S1,
                                      unsigned short* __restrict__ R1) {
    __shared__ float row[D_N];
    int blk  = blockIdx.x;
    int half = blk & 1;
    int c    = (blk >> 1) & (C_N - 1);
    int t    = blk >> 10;
    int h    = half * 256 + threadIdx.x;

    row[threadIdx.x] = inputs[(size_t)c * D_N + threadIdx.x];
    __syncthreads();

    const float* w = W1 + (size_t)t * (2 * D_N) * H_N;
    float s = b1[t * H_N + h];
    float r = 0.f;
#pragma unroll 4
    for (int d = 0; d < D_N; ++d) {
        float x = row[d];
        s += x * w[(size_t)d * H_N + h];
        r += x * w[(size_t)(d + D_N) * H_N + h];
    }
    S1[((size_t)t * C_N + c) * H_N + h] = f2bf(s);
    R1[((size_t)t * C_N + c) * H_N + h] = f2bf(r);
}

// ---------------------------------------------------------------------------
// Kernel 2: pack W2 (T,H,M) fp32 -> bf16 WMMA B-fragments.
// Fragment (t, kt, nt): 32 lanes x 16 halves, lane L<16 -> col nt*16+L,
// K = kt*32 + 0..15 ; lane L+16 -> same col, K = kt*32 + 16..31.
// ---------------------------------------------------------------------------
__global__ void pack_w2(const float* __restrict__ W2,
                        unsigned short* __restrict__ w2p) {
    int idx  = blockIdx.x * 256 + threadIdx.x;       // < 262144
    int h    = idx & 15;
    int lane = (idx >> 4) & 31;
    int nt   = (idx >> 9) & 15;
    int kt   = (idx >> 13) & 15;
    int t    = idx >> 17;
    int n    = nt * 16 + (lane & 15);
    int k    = kt * 32 + ((lane >> 4) << 4) + h;
    w2p[idx] = f2bf(W2[((size_t)t * H_N + k) * M_N + n]);
}

// ---------------------------------------------------------------------------
// Kernel 3: per-edge layer-2 GEMM via bf16 WMMA + scatter-add.
// 64 edges per WG, 8 waves. Wave w owns cols [w*32, w*32+32) (unique B
// fragments per wave -> no intra-WG B redundancy) and loops all 4 edge
// tiles, re-reading A from LDS (cheap).
// ---------------------------------------------------------------------------
__global__ __launch_bounds__(256)
void edge_msgs(const unsigned short* __restrict__ S1,
               const unsigned short* __restrict__ R1,
               const unsigned short* __restrict__ w2p,
               const float* __restrict__ b2,
               const float* __restrict__ rel,        // [E,2]
               const int* __restrict__ send_idx,
               const int* __restrict__ rec_idx,
               float* __restrict__ agg) {
    __shared__ unsigned short hA[64 * H_N];          // 64 KB bf16 hidden tile
    __shared__ int   sendL[64], recL[64];
    __shared__ float relL[T_N][64];
    __shared__ float b2L[T_N * M_N];

    const int tid = threadIdx.x;
    const int e0  = blockIdx.x * 64;

    if (tid < 64) { sendL[tid] = send_idx[e0 + tid]; recL[tid] = rec_idx[e0 + tid]; }
    if (tid < 128) { int e = tid >> 1, t = tid & 1; relL[t][e] = rel[(size_t)(e0 + e) * 2 + t]; }
    b2L[tid]       = b2[tid];
    b2L[256 + tid] = b2[256 + tid];

    const int lane    = tid & 31;
    const int wave    = tid >> 5;
    const int colBase = wave * 32;            // 32-col strip per wave
    const int mSel    = (lane >> 4) & 1;      // lane half
    const int rA      = lane & 15;
    const int n       = lane & 15;

    const v8f vzero = {};
    v8f acc[8];                               // [et*2 + ct]
#pragma unroll
    for (int i = 0; i < 8; ++i) acc[i] = vzero;

    for (int t = 0; t < T_N; ++t) {
        __syncthreads();                      // hA safe to overwrite
        // build hidden tile: relu(S1[send] + R1[rec]) -> bf16 LDS
#pragma unroll 4
        for (int i = 0; i < 128; ++i) {
            int v = i * 256 + tid;
            int e = v >> 9;
            int k = v & (H_N - 1);
            float x = bf2f(S1[((size_t)t * C_N + sendL[e]) * H_N + k])
                    + bf2f(R1[((size_t)t * C_N + recL[e]) * H_N + k]);
            hA[(e << 9) + k] = f2bf(x > 0.f ? x : 0.f);
        }
        __syncthreads();

        v8f accT[8];
#pragma unroll
        for (int i = 0; i < 8; ++i) accT[i] = vzero;

        const unsigned short* wbase = w2p + ((size_t)t << 17);
        for (int kt = 0; kt < 16; ++kt) {
            // two B fragments (this wave's two 16-col tiles), unique per wave
            union { uint4 q[2]; v16bf v; } B0, B1;
            const uint4* bp0 = (const uint4*)(wbase +
                    (((size_t)(kt * 16 + wave * 2 + 0) * 32 + lane) << 4));
            const uint4* bp1 = (const uint4*)(wbase +
                    (((size_t)(kt * 16 + wave * 2 + 1) * 32 + lane) << 4));
            B0.q[0] = bp0[0]; B0.q[1] = bp0[1];
            B1.q[0] = bp1[0]; B1.q[1] = bp1[1];
#pragma unroll
            for (int et = 0; et < 4; ++et) {
                // A fragment from LDS (contiguous k order, same as packed B)
                union { uint4 q[2]; v16bf v; } A;
                const uint4* ap = (const uint4*)
                    &hA[((et * 16 + rA) << 9) + kt * 32 + mSel * 16];
                A.q[0] = ap[0];      // K = k0 + mSel*16 + 0..7
                A.q[1] = ap[1];      // K = k0 + mSel*16 + 8..15
                accT[et * 2 + 0] = __builtin_amdgcn_wmma_f32_16x16x32_bf16(
                    false, A.v, false, B0.v, (short)0, accT[et * 2 + 0], false, false);
                accT[et * 2 + 1] = __builtin_amdgcn_wmma_f32_16x16x32_bf16(
                    false, A.v, false, B1.v, (short)0, accT[et * 2 + 1], false, false);
            }
        }

        // bias + relu + rel_type scale, accumulate over edge types
#pragma unroll
        for (int et = 0; et < 4; ++et) {
#pragma unroll
            for (int ct = 0; ct < 2; ++ct) {
                float bb = b2L[t * M_N + colBase + ct * 16 + n];
#pragma unroll
                for (int j = 0; j < 8; ++j) {
                    int m   = j + mSel * 8;   // C layout: VGPR j, lane half
                    float x = accT[et * 2 + ct][j] + bb;
                    x = x > 0.f ? x : 0.f;
                    acc[et * 2 + ct][j] += relL[t][et * 16 + m] * x;
                }
            }
        }
    }

    // scatter-add into agg[rec]
#pragma unroll
    for (int et = 0; et < 4; ++et) {
#pragma unroll
        for (int ct = 0; ct < 2; ++ct) {
            int col = colBase + ct * 16 + n;
#pragma unroll
            for (int j = 0; j < 8; ++j) {
                int m = j + mSel * 8;
                atomicAdd(&agg[(size_t)recL[et * 16 + m] * M_N + col],
                          acc[et * 2 + ct][j]);
            }
        }
    }
}

// ---------------------------------------------------------------------------
// Output MLP layers (tiny, fp32)
// ---------------------------------------------------------------------------
template <int K, int N, bool RELU>
__global__ void dense_layer(const float* __restrict__ X,
                            const float* __restrict__ W,
                            const float* __restrict__ b,
                            float* __restrict__ Y) {
    __shared__ float row[K];
    const int nb  = N / 256;
    const int blk = blockIdx.x;
    const int c   = blk / nb;
    const int o   = (blk % nb) * 256 + threadIdx.x;
    for (int k = threadIdx.x; k < K; k += 256) row[k] = X[(size_t)c * K + k];
    __syncthreads();
    float s = b[o];
#pragma unroll 4
    for (int k = 0; k < K; ++k) s += row[k] * W[(size_t)k * N + o];
    if (RELU) s = s > 0.f ? s : 0.f;
    Y[(size_t)c * N + o] = s;
}

// ---------------------------------------------------------------------------
extern "C" void kernel_launch(void* const* d_in, const int* in_sizes, int n_in,
                              void* d_out, int out_size, void* d_ws, size_t ws_size,
                              hipStream_t stream) {
    const float* inputs = (const float*)d_in[0];
    const float* rel    = (const float*)d_in[1];
    const float* W1     = (const float*)d_in[2];
    const float* b1     = (const float*)d_in[3];
    const float* W2     = (const float*)d_in[4];
    const float* b2     = (const float*)d_in[5];
    const float* Wo1    = (const float*)d_in[6];
    const float* bo1    = (const float*)d_in[7];
    const float* Wo2    = (const float*)d_in[8];
    const float* bo2    = (const float*)d_in[9];
    const float* Wo3    = (const float*)d_in[10];
    const float* bo3    = (const float*)d_in[11];
    const int* send_idx = (const int*)d_in[12];
    const int* rec_idx  = (const int*)d_in[13];
    float* out          = (float*)d_out;

    unsigned short* S1  = (unsigned short*)d_ws;                   // 2*512*512 u16
    unsigned short* R1  = S1 + (size_t)T_N * C_N * H_N;            // 2*512*512 u16
    unsigned short* w2p = R1 + (size_t)T_N * C_N * H_N;            // 262144 u16
    float* agg   = (float*)(w2p + (size_t)T_N * H_N * M_N);        // 512*256 f32
    float* pred1 = agg + (size_t)C_N * M_N;                        // 512*512 f32
    float* pred2 = pred1 + (size_t)C_N * 512;                      // 512*512 f32

    precompute_node_terms<<<T_N * C_N * 2, 256, 0, stream>>>(inputs, W1, b1, S1, R1);
    pack_w2<<<(T_N * H_N * M_N) / 256, 256, 0, stream>>>(W2, w2p);
    hipMemsetAsync(agg, 0, (size_t)C_N * M_N * sizeof(float), stream);
    edge_msgs<<<E_N / 64, 256, 0, stream>>>(S1, R1, w2p, b2, rel, send_idx, rec_idx, agg);
    dense_layer<M_N, 512, true ><<<C_N * 2, 256, 0, stream>>>(agg,   Wo1, bo1, pred1);
    dense_layer<512, 512, true ><<<C_N * 2, 256, 0, stream>>>(pred1, Wo2, bo2, pred2);
    dense_layer<512, D_N, false><<<C_N * 1, 256, 0, stream>>>(pred2, Wo3, bo3, out);
}
